// SiameseRelational_64441689309605
// MI455X (gfx1250) — matrix operand order repaired and checked by reading, not compiled, not run on hardware
//
#include <hip/hip_runtime.h>
#include <hip/hip_bf16.h>

typedef __attribute__((ext_vector_type(8)))  _Float16 v8h;
typedef __attribute__((ext_vector_type(16))) _Float16 v16h;
typedef __attribute__((ext_vector_type(8)))  float    v8f;
typedef __attribute__((ext_vector_type(4)))  float    f4;

#define C_DIM 512   // feature dim (K per GEMM)
#define HID   256   // relation hidden dim (N)
#define BATCH 8
#define L     196   // 14*14
#define LPAD  208   // 13 tiles of 16 rows
#define MT    13
#define NT    16    // 256/16
#define KD    1024  // 2*C rows of W1, stored transposed as [HID][KD]

// ---------------------------------------------------------------------------
// f32 [B,C,L] feature map -> f16 [B,LPAD,C] row-major (rows >= L zero-filled).
// This is the WMMA A-operand-friendly layout: K (=C) contiguous per row.
// ---------------------------------------------------------------------------
__global__ __launch_bounds__(256) void conv_feat(const float* __restrict__ f,
                                                 _Float16* __restrict__ xh,
                                                 int total) {
    int idx = blockIdx.x * 256 + threadIdx.x;
    if (idx >= total) return;
    int c   = idx & (C_DIM - 1);
    int rem = idx >> 9;          // / 512
    int l   = rem % LPAD;
    int b   = rem / LPAD;
    float v = 0.0f;
    if (l < L) v = f[((size_t)b * C_DIM + c) * L + l];
    xh[idx] = (_Float16)v;
}

// ---------------------------------------------------------------------------
// W1 f32 [KD, HID] -> w1t f16 [HID, KD]  (B operand: K contiguous per n-row).
// Columns k<512 feed P1, k>=512 feed P2.
// ---------------------------------------------------------------------------
__global__ __launch_bounds__(256) void conv_w(const float* __restrict__ W1,
                                              _Float16* __restrict__ w1t) {
    int idx = blockIdx.x * 256 + threadIdx.x;   // total = HID*KD
    int k = idx & (KD - 1);
    int n = idx >> 10;
    w1t[idx] = (_Float16)W1[(size_t)k * HID + n];
}

// ---------------------------------------------------------------------------
// One wave per 16x16 tile of P1/P2 = Xh * W1T.  M=LPAD(=13 tiles), N=256,
// K=512, v_wmma_f32_16x16x32_f16, K-loop fully unrolled (16 WMMAs / wave).
//
// Fragment layouts (ISA 7.12.2, wave32, 16-bit):
//   A 16x32:  m = lane&15, half = lane>>4
//             elems 0..7  = A[m, k0 + half*8      .. +7]
//             elems 8..15 = A[m, k0 + 16 + half*8 .. +7]
//   B 32x16:  n = lane&15; elems 0..7 = B[k0 + half*8 .. +7, n], elems 8..15 = B[k0 + 16 + half*8 .. +7, n]
//   C/D f32:  c[r] -> row (r + 8*half), col (lane&15)
// ---------------------------------------------------------------------------
__global__ __launch_bounds__(256) void gemm_wmma(const _Float16* __restrict__ x1h,
                                                 const _Float16* __restrict__ x2h,
                                                 const _Float16* __restrict__ w1t,
                                                 float* __restrict__ p1,
                                                 float* __restrict__ p2) {
    const int lane = threadIdx.x & 31;
    const int wave = threadIdx.x >> 5;
    const int gw   = blockIdx.x * 8 + wave;     // global wave id = tile id

    const int nt  = gw & 15;
    int tmp       = gw >> 4;
    const int mt  = tmp % MT;
    tmp          /= MT;
    const int mat = tmp & 1;                    // 0 -> P1, 1 -> P2
    const int b   = tmp >> 1;

    const int m    = lane & 15;
    const int half = lane >> 4;

    const _Float16* __restrict__ A =
        (mat ? x2h : x1h) + ((size_t)b * LPAD + mt * 16 + m) * C_DIM + half * 8;
    const _Float16* __restrict__ Bp =
        w1t + (size_t)(nt * 16 + m) * KD + mat * C_DIM + half * 8;

    v8f acc = {};
#pragma unroll
    for (int k0 = 0; k0 < C_DIM; k0 += 32) {
        v8h alo = *(const v8h*)(A + k0);        // K = k0+half*8    .. +7
        v8h ahi = *(const v8h*)(A + k0 + 16);   // K = k0+16+half*8 .. +7
        v8h blo = *(const v8h*)(Bp + k0);       // K = k0+half*8    .. +7
        v8h bhi = *(const v8h*)(Bp + k0 + 16);  // K = k0+16+half*8 .. +7
        v16h av = __builtin_shufflevector(alo, ahi, 0,1,2,3,4,5,6,7,8,9,10,11,12,13,14,15);
        v16h bv = __builtin_shufflevector(blo, bhi, 0,1,2,3,4,5,6,7,8,9,10,11,12,13,14,15);
        acc = __builtin_amdgcn_wmma_f32_16x16x32_f16(
            /*neg_a=*/false, av, /*neg_b=*/false, bv,
            /*c_mod=*/(short)0, acc, /*reuse_a=*/false, /*reuse_b=*/false);
    }

    float* __restrict__ P = (mat ? p2 : p1) + (size_t)b * LPAD * HID;
    const int col = nt * 16 + m;
#pragma unroll
    for (int r = 0; r < 8; ++r) {
        P[(size_t)(mt * 16 + r + 8 * half) * HID + col] = acc[r];
    }
}

// ---------------------------------------------------------------------------
// Pairwise relu-dot: block = (b, i); sA[h] = P1[b,i,h] + b1[h] and W2 in LDS;
// thread j streams P2 row j (float4, all L2-resident) and accumulates
// sum_h relu(sA[h] + P2[j,h]) * W2[h].  Block partial -> ws (deterministic).
// ---------------------------------------------------------------------------
__global__ __launch_bounds__(256) void relate(const float* __restrict__ p1,
                                              const float* __restrict__ p2,
                                              const float* __restrict__ b1,
                                              const float* __restrict__ w2,
                                              float* __restrict__ partial) {
    const int b = blockIdx.x / L;
    const int i = blockIdx.x - b * L;
    const int t = threadIdx.x;

    __shared__ float sA[HID];
    __shared__ float sW[HID];
    sA[t] = p1[((size_t)b * LPAD + i) * HID + t] + b1[t];
    sW[t] = w2[t];
    __syncthreads();

    float acc = 0.0f;
    if (t < L) {
        const f4* __restrict__ row = (const f4*)(p2 + ((size_t)b * LPAD + t) * HID);
        const f4* sA4 = (const f4*)sA;
        const f4* sW4 = (const f4*)sW;
#pragma unroll 4
        for (int h = 0; h < HID / 4; ++h) {
            f4 r = row[h];
            f4 a = sA4[h];
            f4 w = sW4[h];
            acc += fmaxf(a.x + r.x, 0.0f) * w.x;
            acc += fmaxf(a.y + r.y, 0.0f) * w.y;
            acc += fmaxf(a.z + r.z, 0.0f) * w.z;
            acc += fmaxf(a.w + r.w, 0.0f) * w.w;
        }
    }

    __shared__ float red[256];
    red[t] = acc;
    __syncthreads();
#pragma unroll
    for (int s = 128; s > 0; s >>= 1) {
        if (t < s) red[t] += red[t + s];
        __syncthreads();
    }
    if (t == 0) partial[blockIdx.x] = red[0];
}

// ---------------------------------------------------------------------------
// Final: out[b] = sum of 196 block partials + b2 * L*L.
// ---------------------------------------------------------------------------
__global__ __launch_bounds__(256) void final_sum(const float* __restrict__ partial,
                                                 const float* __restrict__ b2,
                                                 float* __restrict__ out) {
    const int b = blockIdx.x;
    const int t = threadIdx.x;
    __shared__ float red[256];
    red[t] = (t < L) ? partial[b * L + t] : 0.0f;
    __syncthreads();
#pragma unroll
    for (int s = 128; s > 0; s >>= 1) {
        if (t < s) red[t] += red[t + s];
        __syncthreads();
    }
    if (t == 0) out[b] = red[0] + b2[0] * (float)(L * L);
}

// ---------------------------------------------------------------------------
extern "C" void kernel_launch(void* const* d_in, const int* in_sizes, int n_in,
                              void* d_out, int out_size, void* d_ws, size_t ws_size,
                              hipStream_t stream) {
    const float* f1 = (const float*)d_in[0];   // [B, C, 14, 14]
    const float* f2 = (const float*)d_in[1];
    const float* W1 = (const float*)d_in[2];   // [2C, HID]
    const float* b1 = (const float*)d_in[3];   // [HID]
    const float* W2 = (const float*)d_in[4];   // [HID, 1]
    const float* b2 = (const float*)d_in[5];   // [1]
    float* out = (float*)d_out;                // [B, 1]

    char* ws = (char*)d_ws;
    size_t off = 0;
    _Float16* x1h = (_Float16*)(ws + off); off += (size_t)BATCH * LPAD * C_DIM * 2;
    _Float16* x2h = (_Float16*)(ws + off); off += (size_t)BATCH * LPAD * C_DIM * 2;
    _Float16* w1t = (_Float16*)(ws + off); off += (size_t)HID * KD * 2;
    float* p1     = (float*)(ws + off);    off += (size_t)BATCH * LPAD * HID * 4;
    float* p2     = (float*)(ws + off);    off += (size_t)BATCH * LPAD * HID * 4;
    float* partial= (float*)(ws + off);    off += (size_t)BATCH * L * 4;
    // total workspace use: ~7.35 MB

    const int totalFeat = BATCH * LPAD * C_DIM;                  // 851,968
    conv_feat<<<(totalFeat + 255) / 256, 256, 0, stream>>>(f1, x1h, totalFeat);
    conv_feat<<<(totalFeat + 255) / 256, 256, 0, stream>>>(f2, x2h, totalFeat);
    conv_w<<<(HID * KD) / 256, 256, 0, stream>>>(W1, w1t);       // 1024 blocks
    gemm_wmma<<<(BATCH * 2 * MT * NT) / 8, 256, 0, stream>>>(    // 416 blocks, 8 waves each
        x1h, x2h, w1t, p1, p2);
    relate<<<BATCH * L, 256, 0, stream>>>(p1, p2, b1, W2, partial); // 1568 blocks
    final_sum<<<BATCH, 256, 0, stream>>>(partial, b2, out);
}
